// ConditionalTriangleAttention_20306605376126
// MI455X (gfx1250) — matrix operand order repaired and checked by reading, not compiled, not run on hardware
//
#include <hip/hip_runtime.h>

typedef __attribute__((ext_vector_type(16))) __bf16 v16bf;
typedef __attribute__((ext_vector_type(8)))  float  v8f;

union BFrag { v16bf v; unsigned short u[16]; };

static constexpr int Bb = 2, Nn = 40, HID = 256, NH = 8, CD = 128, HD = 32;
static constexpr int NN  = Nn * Nn;     // 1600
static constexpr int NNP = 1664;        // padded kl stride for VT (zeros in [1600,1664))
static constexpr int Mrows = Bb * NN;   // 3200

// ---------------- workspace layout (bytes, all 16B aligned) ----------------
static constexpr size_t OFF_CONDP = 0;                          // [B][HID] f32
static constexpr size_t OFF_CONDG = OFF_CONDP + 2048;           // [B][HID] f32
static constexpr size_t OFF_WT    = OFF_CONDG + 2048;           // 5x [256][256] bf16, transposed [n][k]
static constexpr size_t WT_SZ     = (size_t)HID * HID * 2;      // 131072
static constexpr size_t OFF_Q     = OFF_WT + 5 * WT_SZ;         // [B][NH][1600][32] bf16
static constexpr size_t QK_SZ     = (size_t)Mrows * HID * 2;    // 1638400
static constexpr size_t OFF_K     = OFF_Q + QK_SZ;
static constexpr size_t OFF_V     = OFF_K + QK_SZ;              // VT: [B][NH][32][1664] bf16
static constexpr size_t VT_SZ     = (size_t)Bb * NH * HD * NNP * 2;  // 1703936
static constexpr size_t OFF_ATT   = OFF_V + VT_SZ;              // [3200][256] bf16

#define DEVINL __device__ __forceinline__

DEVINL unsigned short f2bf(float f) {
  union { float f; unsigned int u; } x; x.f = f;
  unsigned int r = x.u + 0x7FFFu + ((x.u >> 16) & 1u);
  return (unsigned short)(r >> 16);
}
DEVINL float bf2f(unsigned short h) {
  union { unsigned int u; float f; } x; x.u = ((unsigned int)h) << 16;
  return x.f;
}

DEVINL v8f wmma_bf16(v16bf a, v16bf b, v8f c) {
  return __builtin_amdgcn_wmma_f32_16x16x32_bf16(false, a, false, b, (short)0, c, false, false);
}

// A fragment: 16x32 tile from row-major src (ld elems/row).
DEVINL v16bf load_a(const unsigned short* src, int ld) {
  int lane = threadIdx.x & 31;
  int g = lane >> 4, m = lane & 15;
  const unsigned short* row = src + m * ld;
  BFrag f;
#pragma unroll
  for (int p = 0; p < 8; ++p) {
    int kb = p * 2 + g * 16;
    f.u[2 * p]     = row[kb];
    f.u[2 * p + 1] = row[kb + 1];
  }
  return f.v;
}

// B fragment with B[k][n] = src[n][k] (k contiguous): 32B/lane -> b128 pair
DEVINL v16bf load_bT(const unsigned short* src, int ld, int n0, int k0) {
  int lane = threadIdx.x & 31;
  int g = lane >> 4, r = lane & 15;
  const unsigned short* p = src + (size_t)(n0 + r) * ld + k0 + g * 16;
  BFrag f;
#pragma unroll
  for (int i = 0; i < 16; ++i) f.u[i] = p[i];
  return f.v;
}

DEVINL v16bf load_bT_clamp(const unsigned short* src, int ld, int n0, int nmax, int k0) {
  int lane = threadIdx.x & 31;
  int g = lane >> 4, r = lane & 15;
  int n = n0 + r; if (n > nmax) n = nmax;
  const unsigned short* p = src + (size_t)n * ld + k0 + g * 16;
  BFrag f;
#pragma unroll
  for (int i = 0; i < 16; ++i) f.u[i] = p[i];
  return f.v;
}

// pipelined 16x256 @ 256x16 tile GEMM: prefetch next B fragment before each WMMA
DEVINL v8f gemm_tile(const unsigned short* sTile, const unsigned short* wt, int nt) {
  v8f acc = {0.f, 0.f, 0.f, 0.f, 0.f, 0.f, 0.f, 0.f};
  v16bf bn = load_bT(wt, HID, nt * 16, 0);
#pragma unroll
  for (int kc = 0; kc < 8; ++kc) {
    v16bf bc = bn;
    if (kc < 7) bn = load_bT(wt, HID, nt * 16, (kc + 1) * 32);
    v16bf a = load_a(sTile + kc * 32, HID);
    acc = wmma_bf16(a, bc, acc);
  }
  return acc;
}

// ---------------- async global->LDS tile copy (gfx1250 path, guarded) ----------------
#if defined(__has_builtin)
# if __has_builtin(__builtin_amdgcn_global_load_async_to_lds_b128)
#  define HAVE_ASYNC_LDS 1
# endif
#endif
#ifndef HAVE_ASYNC_LDS
# define HAVE_ASYNC_LDS 0
#endif

typedef int i4v __attribute__((vector_size(16)));
typedef __attribute__((address_space(1))) i4v gas_i4;   // global int4*
typedef __attribute__((address_space(3))) i4v las_i4;   // LDS int4*

DEVINL void tile_copy_8192(const unsigned short* gsrc, unsigned short* ldst) {
#if HAVE_ASYNC_LDS
  int t = threadIdx.x;                      // 256 threads x 32B
  const unsigned short* s = gsrc + t * 16;
  unsigned short* d = ldst + t * 16;
  __builtin_amdgcn_global_load_async_to_lds_b128(
      (gas_i4*)(unsigned long long)s,
      (las_i4*)(unsigned int)(unsigned long long)d, 0, 0);
  __builtin_amdgcn_global_load_async_to_lds_b128(
      (gas_i4*)(unsigned long long)(s + 8),
      (las_i4*)(unsigned int)(unsigned long long)(d + 8), 0, 0);
#else
  const unsigned int* src = (const unsigned int*)gsrc;
  unsigned int* dst = (unsigned int*)ldst;
  for (int i = threadIdx.x; i < 2048; i += 256) dst[i] = src[i];
#endif
}
DEVINL void tile_copy_wait() {
#if HAVE_ASYNC_LDS
# if __has_builtin(__builtin_amdgcn_s_wait_asynccnt)
  __builtin_amdgcn_s_wait_asynccnt(0);
# else
  asm volatile("s_wait_asynccnt 0x0" ::: "memory");
# endif
#endif
  __syncthreads();
}

// ---------------- kernels ----------------

__global__ void k_prep_cond(const float* cond, const float* Wcp, const float* bcp,
                            const float* Wcg, const float* bcg,
                            float* condp, float* condg) {
  int t = threadIdx.x;              // 512 = B*HID
  int b = t >> 8, c = t & 255;
  float sp = bcp[c], sg = bcg[c];
  for (int j = 0; j < CD; ++j) {
    float cv = cond[b * CD + j];
    sp += cv * Wcp[j * HID + c];
    sg += cv * Wcg[j * HID + c];
  }
  condp[b * HID + c] = sp;
  condg[b * HID + c] = 1.f / (1.f + __expf(-sg));
}

__global__ void k_prep_w(const float* Wq, const float* Wk, const float* Wv,
                         const float* Wtg, const float* Wo, unsigned short* wt) {
  int idx = blockIdx.x * blockDim.x + threadIdx.x;
  if (idx >= 5 * HID * HID) return;
  int w = idx >> 16, rem = idx & 65535;
  int k = rem >> 8, n = rem & 255;
  const float* src = (w == 0) ? Wq : (w == 1) ? Wk : (w == 2) ? Wv : (w == 3) ? Wtg : Wo;
  wt[(size_t)w * HID * HID + (size_t)n * HID + k] = f2bf(src[k * HID + n]);
}

__global__ void k_zero_vt(unsigned short* VT) {   // zero the kl-pad columns of VT
  int idx = blockIdx.x * blockDim.x + threadIdx.x;
  if (idx >= Bb * NH * HD * (NNP - NN)) return;   // 512 rows x 64 pad cols
  int pr = idx >> 6, c = idx & 63;
  VT[(size_t)pr * NNP + NN + c] = 0;
}

// fused: cf = ef*gate+proj (elementwise, into LDS) then QKV projection GEMMs
__global__ __launch_bounds__(256) void k_qkv(const float* ef,
    const float* condp, const float* condg,
    const unsigned short* wtq, const unsigned short* wtk, const unsigned short* wtv,
    const float* bq, const float* bk, const float* bv,
    unsigned short* Qo, unsigned short* Ko, unsigned short* VTo) {
  __shared__ unsigned short sCF[16 * HID];
  int mt = blockIdx.x;
  int bidx = mt / 100;                       // 100 row-tiles per batch
  const float* efb = ef + (size_t)mt * 16 * HID;
  for (int i = threadIdx.x; i < 16 * HID; i += 256) {
    int c = i & 255;
    sCF[i] = f2bf(efb[i] * condg[bidx * HID + c] + condp[bidx * HID + c]);
  }
  __syncthreads();
  int wave = threadIdx.x >> 5, lane = threadIdx.x & 31;
  int g = lane >> 4, r = lane & 15;
  for (int tci = 0; tci < 6; ++tci) {        // 48 tile-columns / 8 waves
    int tc = wave * 6 + tci;
    int sel = tc >> 4, nt = tc & 15;
    const unsigned short* wt = (sel == 0) ? wtq : (sel == 1) ? wtk : wtv;
    v8f acc = gemm_tile(sCF, wt, nt);
    int col = nt * 16 + r;
    const float* bias = (sel == 0) ? bq : (sel == 1) ? bk : bv;
    float bval = bias[col];
    float scale = (sel == 0) ? 0.17677669529663687f : 1.0f;   // Q pre-scaled by 1/sqrt(HD)
    int h = col >> 5, d = col & 31;
#pragma unroll
    for (int u = 0; u < 8; ++u) {
      int row = mt * 16 + u + g * 8;
      int ij = row % NN;
      unsigned short val = f2bf((acc[u] + bval) * scale);
      if (sel == 2)      // V transposed: VT[(b*NH+h)*HD + d][kl]
        VTo[((size_t)(bidx * NH + h) * HD + d) * NNP + ij] = val;
      else if (sel == 0)
        Qo[((size_t)(bidx * NH + h) * NN + ij) * HD + d] = val;
      else
        Ko[((size_t)(bidx * NH + h) * NN + ij) * HD + d] = val;
    }
  }
}

__global__ __launch_bounds__(32) void k_attn(const unsigned short* Q, const unsigned short* K,
                                             const unsigned short* VT, const unsigned char* mask,
                                             unsigned short* att) {
  __shared__ float S[16 * 48];            // score tile: 16 queries x 48 padded keys
  __shared__ unsigned short P[16 * 64];   // probs bf16, padded to 64 for 2 K-chunks
  __shared__ float Sred[32];              // softmax cross-half reduction
  int qt = blockIdx.x, h = blockIdx.y, b = blockIdx.z;
  int lane = threadIdx.x;
  int g = lane >> 4, r = lane & 15;
  size_t bh = (size_t)(b * NH + h);
  const unsigned short* Qb  = Q  + (bh * NN + (size_t)qt * 16) * HD;
  const unsigned short* Kb  = K  + bh * NN * HD;
  const unsigned short* VTp = VT + bh * HD * NNP;
  const unsigned char* mb = mask + b * NN;

  v16bf aQ = load_a(Qb, HD);              // 16x32 query tile, reused for all k

  if (lane < 16)
    for (int c = 48; c < 64; ++c) P[lane * 64 + c] = 0;   // permanent zero pad
  v8f acc0 = {0.f, 0.f, 0.f, 0.f, 0.f, 0.f, 0.f, 0.f};
  v8f acc1 = {0.f, 0.f, 0.f, 0.f, 0.f, 0.f, 0.f, 0.f};

  for (int k = 0; k < Nn; ++k) {
    const unsigned short* Kk  = Kb + (size_t)k * Nn * HD;
    const unsigned short* VTk = VTp + (size_t)k * Nn;     // column offset into padded rows
    const unsigned char* mk = mb + k * Nn;
    // ---- scores: issue all 3 B fragments, then 3 WMMAs
    v16bf bK0 = load_bT_clamp(Kk, HD, 0,  Nn - 1, 0);
    v16bf bK1 = load_bT_clamp(Kk, HD, 16, Nn - 1, 0);
    v16bf bK2 = load_bT_clamp(Kk, HD, 32, Nn - 1, 0);
    v8f z = {0.f, 0.f, 0.f, 0.f, 0.f, 0.f, 0.f, 0.f};
    v8f cs[3];
    cs[0] = wmma_bf16(aQ, bK0, z);
    cs[1] = wmma_bf16(aQ, bK1, z);
    cs[2] = wmma_bf16(aQ, bK2, z);
#pragma unroll
    for (int t = 0; t < 3; ++t) {
      int l = t * 16 + r;
      bool valid = (l < Nn) ? (mk[l] != 0) : false;
#pragma unroll
      for (int u = 0; u < 8; ++u)
        S[(u + g * 8) * 48 + l] = valid ? cs[t][u] : -__builtin_inff();
    }
    __syncthreads();
    // ---- per-row softmax over l, both lane-halves cooperate (24 cols each)
    {
      int rrow = lane & 15, half = lane >> 4, c0 = half * 24;
      float mx = -__builtin_inff();
      for (int c = 0; c < 24; ++c) { float s = S[rrow * 48 + c0 + c]; mx = s > mx ? s : mx; }
      Sred[half * 16 + rrow] = mx;
      __syncthreads();
      float gmx = fmaxf(Sred[rrow], Sred[16 + rrow]);
      float sum = 0.f;
      for (int c = 0; c < 24; ++c) sum += __expf(S[rrow * 48 + c0 + c] - gmx);
      __syncthreads();
      Sred[half * 16 + rrow] = sum;
      __syncthreads();
      float inv = 1.f / (Sred[rrow] + Sred[16 + rrow]);
      for (int c = 0; c < 24; ++c) {
        int l = c0 + c;                       // pads are -inf -> exp = 0
        P[rrow * 64 + l] = f2bf(__expf(S[rrow * 48 + l] - gmx) * inv);
      }
    }
    __syncthreads();
    // ---- P(16x64) @ V(64x32 via VT, zero-padded): contiguous b128 B-frags
    v16bf aP0 = load_a(P, 64);
    v16bf aP1 = load_a(P + 32, 64);
    v16bf b00 = load_bT(VTk, NNP, 0,  0);
    v16bf b01 = load_bT(VTk, NNP, 16, 0);
    v16bf b10 = load_bT(VTk, NNP, 0,  32);
    v16bf b11 = load_bT(VTk, NNP, 16, 32);
    acc0 = wmma_bf16(aP0, b00, acc0);
    acc1 = wmma_bf16(aP0, b01, acc1);
    acc0 = wmma_bf16(aP1, b10, acc0);
    acc1 = wmma_bf16(aP1, b11, acc1);
    __syncthreads();
  }
  int rowbase = b * NN + qt * 16;
#pragma unroll
  for (int u = 0; u < 8; ++u) {
    int row = rowbase + u + g * 8;
    att[(size_t)row * HID + h * 32 + r]      = f2bf(acc0[u]);
    att[(size_t)row * HID + h * 32 + 16 + r] = f2bf(acc1[u]);
  }
}

__global__ __launch_bounds__(256) void k_out(const unsigned short* att,
    const unsigned short* wttg, const unsigned short* wto,
    const float* btg, const float* bo, float* out) {
  __shared__ unsigned short sA[16 * HID];
  __shared__ unsigned short sG[16 * HID];
  int mt = blockIdx.x;
  tile_copy_8192(att + (size_t)mt * 16 * HID, sA);   // async global->LDS when available
  tile_copy_wait();
  int wave = threadIdx.x >> 5, lane = threadIdx.x & 31;
  int g = lane >> 4, r = lane & 15;
  // GEMM1: gate = sigmoid(att @ Wtg + btg); gated = att * gate
  for (int ti = 0; ti < 2; ++ti) {
    int nt = wave * 2 + ti;
    v8f acc = gemm_tile(sA, wttg, nt);
    int col = nt * 16 + r;
    float bval = btg[col];
#pragma unroll
    for (int u = 0; u < 8; ++u) {
      int m = u + g * 8;
      float gate = 1.f / (1.f + __expf(-(acc[u] + bval)));
      sG[m * HID + col] = f2bf(bf2f(sA[m * HID + col]) * gate);
    }
  }
  __syncthreads();
  // GEMM2: out = gated @ Wo + bo (f32 output)
  for (int ti = 0; ti < 2; ++ti) {
    int nt = wave * 2 + ti;
    v8f acc = gemm_tile(sG, wto, nt);
    int col = nt * 16 + r;
    float bval = bo[col];
#pragma unroll
    for (int u = 0; u < 8; ++u) {
      int row = mt * 16 + u + g * 8;
      out[(size_t)row * HID + col] = acc[u] + bval;
    }
  }
}

// ---------------- launch ----------------

extern "C" void kernel_launch(void* const* d_in, const int* in_sizes, int n_in,
                              void* d_out, int out_size, void* d_ws, size_t ws_size,
                              hipStream_t stream) {
  const float* ef            = (const float*)d_in[0];
  const unsigned char* emask = (const unsigned char*)d_in[1];
  const float* cond          = (const float*)d_in[2];
  const float* Wq  = (const float*)d_in[3];  const float* bq  = (const float*)d_in[4];
  const float* Wk  = (const float*)d_in[5];  const float* bk  = (const float*)d_in[6];
  const float* Wv  = (const float*)d_in[7];  const float* bv  = (const float*)d_in[8];
  const float* Wcp = (const float*)d_in[9];  const float* bcp = (const float*)d_in[10];
  const float* Wcg = (const float*)d_in[11]; const float* bcg = (const float*)d_in[12];
  // d_in[13]/d_in[14] (Wtb/btb): additive bias is constant along softmax axis -> cancels
  const float* Wtg = (const float*)d_in[15]; const float* btg = (const float*)d_in[16];
  const float* Wo  = (const float*)d_in[17]; const float* bo  = (const float*)d_in[18];

  char* ws = (char*)d_ws;
  float* condp = (float*)(ws + OFF_CONDP);
  float* condg = (float*)(ws + OFF_CONDG);
  unsigned short* wt   = (unsigned short*)(ws + OFF_WT);
  unsigned short* wtq  = wt;
  unsigned short* wtk  = wt + 1 * HID * HID;
  unsigned short* wtv  = wt + 2 * HID * HID;
  unsigned short* wttg = wt + 3 * HID * HID;
  unsigned short* wto  = wt + 4 * HID * HID;
  unsigned short* Qw   = (unsigned short*)(ws + OFF_Q);
  unsigned short* Kw   = (unsigned short*)(ws + OFF_K);
  unsigned short* VTw  = (unsigned short*)(ws + OFF_V);
  unsigned short* attw = (unsigned short*)(ws + OFF_ATT);

  k_prep_cond<<<1, 512, 0, stream>>>(cond, Wcp, bcp, Wcg, bcg, condp, condg);
  k_prep_w<<<(5 * HID * HID + 255) / 256, 256, 0, stream>>>(Wq, Wk, Wv, Wtg, Wo, wt);
  k_zero_vt<<<(Bb * NH * HD * (NNP - NN) + 255) / 256, 256, 0, stream>>>(VTw);
  k_qkv<<<Mrows / 16, 256, 0, stream>>>(ef, condp, condg, wtq, wtk, wtv,
                                        bq, bk, bv, Qw, Kw, VTw);
  k_attn<<<dim3(NN / 16, NH, Bb), 32, 0, stream>>>(Qw, Kw, VTw, emask, attw);
  k_out<<<Mrows / 16, 256, 0, stream>>>(attw, wttg, wto, btg, bo, (float*)d_out);
}
